// NewRelativeEncoderLayer_59871844107115
// MI455X (gfx1250) — matrix operand order repaired and checked by reading, not compiled, not run on hardware
//
#include <hip/hip_runtime.h>
#include <hip/hip_bf16.h>
#include <stdint.h>

// ---------------------------------------------------------------------------
// NewRelativeEncoderLayer for MI455X (gfx1250, wave32, WMMA).
//
// Roofline: ~22 GFLOP total, but the a_k intermediate [4,512,512,64] would be
// 268MB fp32 (~540MB HBM round-trip). We fuse the rel-pos MLP into the score
// kernel so a_k only ever exists as a 64x64 LDS tile. All dense GEMMs use
// v_wmma_f32_16x16x32_bf16 (bf16 in, fp32 accumulate), 64x32 wave tiles
// (8 wmma per 12 b128 loads), with global_prefetch for the next K tile.
// ---------------------------------------------------------------------------

#define BB      4
#define SEQ     512
#define DMODEL  512
#define NHEAD   8
#define DHEAD   64
#define DINNER  2048
#define TOTROWS (BB * SEQ)      // 2048
#define OUT2_ELEMS ((long)BB * SEQ * DMODEL)          // 1,048,576
#define ATTN_ELEMS ((long)BB * NHEAD * SEQ * SEQ)     // 8,388,608

typedef __attribute__((ext_vector_type(16))) __bf16 v16bf;
typedef __attribute__((ext_vector_type(8)))  float  v8f;

union Frag16 { v16bf v; unsigned short u[16]; };
union Acc8   { v8f   v; float f[8]; };

__device__ __forceinline__ unsigned short f2bf(float f) {
  unsigned int u = __builtin_bit_cast(unsigned int, f);
  u += 0x7FFFu + ((u >> 16) & 1u);          // round-to-nearest-even
  return (unsigned short)(u >> 16);
}

__device__ __forceinline__ void load8(unsigned short* dst, const unsigned short* src) {
  union { uint4 q; unsigned short s[8]; } u;
  u.q = *reinterpret_cast<const uint4*>(src);
#pragma unroll
  for (int i = 0; i < 8; ++i) dst[i] = u.s[i];
}

// --------------------------- fp32 -> bf16 convert ---------------------------
__global__ void cvt_bf16_kernel(const float* __restrict__ in,
                                unsigned short* __restrict__ out, long n) {
  long i = (long)blockIdx.x * blockDim.x + threadIdx.x;
  if (i < n) out[i] = f2bf(in[i]);
}

// --------------------------- V transpose (per b,h) ---------------------------
// vb [b*512+i][h*64+d]  ->  vtb [(b*8+h)*64+d][i]   (WMMA "Bt" layout for attn@V)
__global__ void vtrans_kernel(const unsigned short* __restrict__ vb,
                              unsigned short* __restrict__ vtb) {
  const long idx = (long)blockIdx.x * 256 + threadIdx.x;   // 32*64*512 total
  const int  i = (int)(idx & (SEQ - 1));
  const long r = idx >> 9;
  const int  d = (int)(r & (DHEAD - 1));
  const int  z = (int)(r >> 6);
  const int  b = z >> 3, h = z & 7;
  vtb[idx] = vb[(long)(b * SEQ + i) * DMODEL + h * DHEAD + d];
}

// ------------------------------- WMMA GEMM ----------------------------------
// C[M,N] = A[M,K] (bf16,row-major) @ Bt[N,K]^T (bf16; Bt row n holds column n)
// 256 threads = 8 waves as 4(m) x 2(n); wave tile 64x32 (4x2 wmma),
// block tile 256x64.  8 wmma per 12 b128 loads per K step of 32.
// MODE: 0 store fp32 | 1 store bf16 | 2 bias+relu->bf16 | 3 +res->fp32
//       4 bias+res->fp32
template <int MODE>
__global__ __launch_bounds__(256) void gemm_bf16_kernel(
    const unsigned short* __restrict__ A, int lda, long aStride,
    const unsigned short* __restrict__ Bt, int ldb, long bStride,
    float* __restrict__ Cf, unsigned short* __restrict__ Cb, int ldc,
    int cDiv, long cStrideOuter, long cStrideInner,
    int M, int N, int K,
    const float* __restrict__ bias, const float* __restrict__ res)
{
  const int z = blockIdx.z;
  A  += (long)z * aStride;
  Bt += (long)z * bStride;
  const long cOff = (long)(z / cDiv) * cStrideOuter + (long)(z % cDiv) * cStrideInner;

  const int lane = threadIdx.x & 31;
  const int w    = threadIdx.x >> 5;
  const int wm = w & 3, wn = w >> 2;
  const int m0 = blockIdx.y * 256 + wm * 64;
  const int n0 = blockIdx.x * 64 + wn * 32;
  if (m0 >= M || n0 >= N) return;                 // wave-uniform exit
  const int l16 = lane & 15, lh = lane >> 4;

  const v8f zero = {0.f, 0.f, 0.f, 0.f, 0.f, 0.f, 0.f, 0.f};
  v8f acc[4][2];
#pragma unroll
  for (int i = 0; i < 4; ++i)
#pragma unroll
    for (int j = 0; j < 2; ++j) acc[i][j] = zero;

  for (int k0 = 0; k0 < K; k0 += 32) {
    Frag16 af[4], bf[2];
#pragma unroll
    for (int t = 0; t < 4; ++t) {
      // A frag (16x32): lane m = l16; K chunks {lh*8.., 16+lh*8..}
      const unsigned short* ap = A + (long)(m0 + t * 16 + l16) * lda + k0;
      load8(&af[t].u[0], ap + lh * 8);
      load8(&af[t].u[8], ap + 16 + lh * 8);
      __builtin_prefetch(ap + 32, 0, 3);         // next K tile (global_prefetch)
    }
#pragma unroll
    for (int t = 0; t < 2; ++t) {
      // B frag (32x16): lane n = l16; K chunks {lh*8.., 16+lh*8..}
      const unsigned short* bp = Bt + (long)(n0 + t * 16 + l16) * ldb + k0;
      load8(&bf[t].u[0], bp + lh * 8);
      load8(&bf[t].u[8], bp + 16 + lh * 8);
      __builtin_prefetch(bp + 32, 0, 3);
    }
#pragma unroll
    for (int i = 0; i < 4; ++i)
#pragma unroll
      for (int j = 0; j < 2; ++j)
        acc[i][j] = __builtin_amdgcn_wmma_f32_16x16x32_bf16(
            false, af[i].v, false, bf[j].v, (short)0, acc[i][j], false, false);
  }

#pragma unroll
  for (int i = 0; i < 4; ++i)
#pragma unroll
    for (int j = 0; j < 2; ++j) {
      Acc8 cu; cu.v = acc[i][j];
      const int n = n0 + j * 16 + l16;
      const float bv = (MODE == 2 || MODE == 4) ? bias[n] : 0.0f;
#pragma unroll
      for (int r = 0; r < 8; ++r) {
        const int  m   = m0 + i * 16 + lh * 8 + r;   // C/D: row = lh*8+r
        const long idx = cOff + (long)m * ldc + n;
        float v = cu.f[r] + bv;
        if (MODE == 3 || MODE == 4) v += res[idx];
        if (MODE == 2) v = fmaxf(v, 0.0f);
        if (MODE == 0 || MODE == 3 || MODE == 4) Cf[idx] = v;
        else                                     Cb[idx] = f2bf(v);
      }
    }
}

// ----------------- fused rel-pos MLP + tensor-product score ------------------
// One workgroup per (b, j).  a_k[j, i-chunk, :] built in LDS:
//   p1 = pos_mat[b,j,i,:] @ rp_w1^T + b1      (K=4, VALU)
//   ak = p1 @ rp_w2^T + b2                    (WMMA bf16, 64x64 tile)
//   raw[b,h,j,i] = (sum_d q[i,d]*ak[i,d]*k[j,d]) / sqrt(64)
__global__ __launch_bounds__(256) void score_kernel(
    const float* __restrict__ pos, const float* __restrict__ rp_w1,
    const float* __restrict__ rp_b1, const unsigned short* __restrict__ rpw2b,
    const float* __restrict__ rp_b2, const float* __restrict__ Qf,
    const float* __restrict__ Kf, float* __restrict__ attn_raw)
{
  const int j = blockIdx.x;
  const int b = blockIdx.y;
  __shared__ unsigned short p1_s[64][72];   // bf16, padded (144B rows)
  __shared__ float          ak_s[64][68];   // fp32, padded
  __shared__ float          kj_s[DMODEL];

  const int t = threadIdx.x;
  for (int c = t; c < DMODEL; c += 256)
    kj_s[c] = Kf[(long)(b * SEQ + j) * DMODEL + c];

  const int lane = t & 31, w = t >> 5;
  const int l16 = lane & 15, lh = lane >> 4;
  const int iL = t >> 2, part = t & 3;
  const float invT = 0.125f;                 // 1/sqrt(64)
  const v8f zero = {0.f, 0.f, 0.f, 0.f, 0.f, 0.f, 0.f, 0.f};

  for (int i0 = 0; i0 < SEQ; i0 += 64) {
    __syncthreads();
    // --- p1 (64 x 64), K=4 VALU ---
    for (int idx = t; idx < 64 * 64; idx += 256) {
      const int i = idx >> 6, e = idx & 63;
      const float4 pv = *reinterpret_cast<const float4*>(
          pos + ((long)(b * SEQ + j) * SEQ + i0 + i) * 4);
      const float4 wv = *reinterpret_cast<const float4*>(rp_w1 + e * 4);
      p1_s[i][e] = f2bf(rp_b1[e] + pv.x * wv.x + pv.y * wv.y +
                        pv.z * wv.z + pv.w * wv.w);
    }
    __syncthreads();
    // --- ak = p1 @ rp_w2^T + b2 via WMMA (16 tiles, 2 per wave) ---
    const int it = w & 3;
#pragma unroll
    for (int q2 = 0; q2 < 2; ++q2) {
      const int dt = (w >> 2) * 2 + q2;
      v8f acc = zero;
#pragma unroll
      for (int ks = 0; ks < 2; ++ks) {
        Frag16 a, bb;
        const unsigned short* ap = &p1_s[it * 16 + l16][ks * 32];
        load8(&a.u[0], ap + lh * 8);
        load8(&a.u[8], ap + 16 + lh * 8);
        const unsigned short* bp = rpw2b + (dt * 16 + l16) * 64 + ks * 32;
        load8(&bb.u[0], bp + lh * 8);
        load8(&bb.u[8], bp + 16 + lh * 8);
        acc = __builtin_amdgcn_wmma_f32_16x16x32_bf16(
            false, a.v, false, bb.v, (short)0, acc, false, false);
      }
      Acc8 cu; cu.v = acc;
      const float b2v = rp_b2[dt * 16 + l16];
#pragma unroll
      for (int r = 0; r < 8; ++r)
        ak_s[it * 16 + lh * 8 + r][dt * 16 + l16] = cu.f[r] + b2v;
    }
    __syncthreads();
    // --- scores: 4 lanes/row partial sums, shfl-combine ---
    const long qrow = (long)(b * SEQ + i0 + iL) * DMODEL;
    const int dbase = part * 16;
#pragma unroll
    for (int h = 0; h < NHEAD; ++h) {
      float s = 0.f;
#pragma unroll
      for (int dd = 0; dd < 16; ++dd)
        s += Qf[qrow + h * DHEAD + dbase + dd] * ak_s[iL][dbase + dd] *
             kj_s[h * DHEAD + dbase + dd];
      s += __shfl_xor(s, 1, 32);
      s += __shfl_xor(s, 2, 32);
      if (part == 0)
        attn_raw[((long)(b * NHEAD + h) * SEQ + j) * SEQ + i0 + iL] = s * invT;
    }
  }
}

// ------------------------------ row softmax ---------------------------------
// one block per row of 512; writes normalized fp32 (in place) + bf16 copy
__global__ __launch_bounds__(256) void softmax_kernel(float* __restrict__ attn,
                                                      unsigned short* __restrict__ attnb)
{
  const long row = blockIdx.x;
  const int t = threadIdx.x;
  __shared__ float red[256];
  float* x = attn + row * SEQ;
  const float v0 = x[t], v1 = x[t + 256];
  red[t] = fmaxf(v0, v1);
  __syncthreads();
  for (int o = 128; o > 0; o >>= 1) {
    if (t < o) red[t] = fmaxf(red[t], red[t + o]);
    __syncthreads();
  }
  const float mx = red[0];
  __syncthreads();
  const float e0 = __expf(v0 - mx), e1 = __expf(v1 - mx);
  red[t] = e0 + e1;
  __syncthreads();
  for (int o = 128; o > 0; o >>= 1) {
    if (t < o) red[t] += red[t + o];
    __syncthreads();
  }
  const float inv = 1.0f / red[0];
  const float y0 = e0 * inv, y1 = e1 * inv;
  x[t] = y0; x[t + 256] = y1;
  attnb[row * SEQ + t]       = f2bf(y0);
  attnb[row * SEQ + t + 256] = f2bf(y1);
}

// ------------------------------- layer norm ---------------------------------
__global__ __launch_bounds__(256) void layernorm_kernel(
    const float* __restrict__ in, const float* __restrict__ g,
    const float* __restrict__ bt, float* __restrict__ outF,
    unsigned short* __restrict__ outB)
{
  const long row = blockIdx.x;
  const int t = threadIdx.x;
  __shared__ float red[256];
  const float* x = in + row * DMODEL;
  const float v0 = x[t], v1 = x[t + 256];
  red[t] = v0 + v1;
  __syncthreads();
  for (int o = 128; o > 0; o >>= 1) {
    if (t < o) red[t] += red[t + o];
    __syncthreads();
  }
  const float mu = red[0] * (1.0f / DMODEL);
  __syncthreads();
  const float d0 = v0 - mu, d1 = v1 - mu;
  red[t] = d0 * d0 + d1 * d1;
  __syncthreads();
  for (int o = 128; o > 0; o >>= 1) {
    if (t < o) red[t] += red[t + o];
    __syncthreads();
  }
  const float inv = rsqrtf(red[0] * (1.0f / DMODEL) + 1e-6f);
  const float y0 = d0 * inv * g[t] + bt[t];
  const float y1 = d1 * inv * g[t + 256] + bt[t + 256];
  if (outF) { outF[row * DMODEL + t] = y0; outF[row * DMODEL + t + 256] = y1; }
  if (outB) { outB[row * DMODEL + t] = f2bf(y0); outB[row * DMODEL + t + 256] = f2bf(y1); }
}

// ----------------------------------------------------------------------------
extern "C" void kernel_launch(void* const* d_in, const int* in_sizes, int n_in,
                              void* d_out, int out_size, void* d_ws, size_t ws_size,
                              hipStream_t stream) {
  (void)in_sizes; (void)n_in; (void)out_size; (void)ws_size;
  const float* enc    = (const float*)d_in[0];
  const float* pos    = (const float*)d_in[1];
  const float* w_qs   = (const float*)d_in[2];
  const float* w_ks   = (const float*)d_in[3];
  const float* w_vs   = (const float*)d_in[4];
  const float* w_fc   = (const float*)d_in[5];
  const float* rp_w1  = (const float*)d_in[6];
  const float* rp_b1  = (const float*)d_in[7];
  const float* rp_w2  = (const float*)d_in[8];
  const float* rp_b2  = (const float*)d_in[9];
  const float* ln1_g  = (const float*)d_in[10];
  const float* ln1_b  = (const float*)d_in[11];
  const float* ln2_g  = (const float*)d_in[12];
  const float* ln2_b  = (const float*)d_in[13];
  const float* ffn_w1 = (const float*)d_in[14];
  const float* ffn_b1 = (const float*)d_in[15];
  const float* ffn_w2 = (const float*)d_in[16];
  const float* ffn_b2 = (const float*)d_in[17];

  float* out2 = (float*)d_out;                 // [4,512,512]
  float* attn = (float*)d_out + OUT2_ELEMS;    // [4,8,512,512]

  // ---- workspace carve-out ----
  char* cur = (char*)d_ws;
  auto alloc = [&](size_t bytes) {
    char* p = cur; cur += (bytes + 255) & ~(size_t)255; return p;
  };
  unsigned short* xb     = (unsigned short*)alloc((size_t)TOTROWS * DMODEL * 2);
  unsigned short* wqb    = (unsigned short*)alloc((size_t)DMODEL * DMODEL * 2);
  unsigned short* wkb    = (unsigned short*)alloc((size_t)DMODEL * DMODEL * 2);
  unsigned short* wvb    = (unsigned short*)alloc((size_t)DMODEL * DMODEL * 2);
  unsigned short* wfcb   = (unsigned short*)alloc((size_t)DMODEL * DMODEL * 2);
  unsigned short* w1b    = (unsigned short*)alloc((size_t)DINNER * DMODEL * 2);
  unsigned short* w2b    = (unsigned short*)alloc((size_t)DMODEL * DINNER * 2);
  unsigned short* rpw2b  = (unsigned short*)alloc((size_t)DHEAD * DHEAD * 2);
  float*          Qf     = (float*)alloc((size_t)TOTROWS * DMODEL * 4);
  float*          Kf     = (float*)alloc((size_t)TOTROWS * DMODEL * 4);
  unsigned short* vb     = (unsigned short*)alloc((size_t)TOTROWS * DMODEL * 2);
  unsigned short* vtb    = (unsigned short*)alloc((size_t)TOTROWS * DMODEL * 2);
  unsigned short* attnb  = (unsigned short*)alloc((size_t)ATTN_ELEMS * 2);
  unsigned short* outh_b = (unsigned short*)alloc((size_t)TOTROWS * DMODEL * 2);
  float*          tmp1   = (float*)alloc((size_t)TOTROWS * DMODEL * 4);
  float*          out1f  = (float*)alloc((size_t)TOTROWS * DMODEL * 4);
  unsigned short* out1b  = (unsigned short*)alloc((size_t)TOTROWS * DMODEL * 2);
  unsigned short* Hb     = (unsigned short*)alloc((size_t)TOTROWS * DINNER * 2);
  float*          tmp2   = (float*)alloc((size_t)TOTROWS * DMODEL * 4);

  auto cvt = [&](const float* in, unsigned short* out, long n) {
    cvt_bf16_kernel<<<dim3((unsigned)((n + 255) / 256)), 256, 0, stream>>>(in, out, n);
  };
  // fp32 -> bf16 conversions (weights stay in [out,in] = WMMA "Bt" layout)
  cvt(enc,    xb,    (long)TOTROWS * DMODEL);
  cvt(w_qs,   wqb,   (long)DMODEL * DMODEL);
  cvt(w_ks,   wkb,   (long)DMODEL * DMODEL);
  cvt(w_vs,   wvb,   (long)DMODEL * DMODEL);
  cvt(w_fc,   wfcb,  (long)DMODEL * DMODEL);
  cvt(ffn_w1, w1b,   (long)DINNER * DMODEL);
  cvt(ffn_w2, w2b,   (long)DMODEL * DINNER);
  cvt(rp_w2,  rpw2b, (long)DHEAD * DHEAD);

  auto gx = [](int N) { return (unsigned)((N + 63) / 64); };
  auto gy = [](int M) { return (unsigned)((M + 255) / 256); };

  // Q = X @ Wq^T (fp32), K = X @ Wk^T (fp32), V = X @ Wv^T (bf16)
  gemm_bf16_kernel<0><<<dim3(gx(DMODEL), gy(TOTROWS), 1), 256, 0, stream>>>(
      xb, DMODEL, 0, wqb, DMODEL, 0, Qf, nullptr, DMODEL, 1, 0, 0,
      TOTROWS, DMODEL, DMODEL, nullptr, nullptr);
  gemm_bf16_kernel<0><<<dim3(gx(DMODEL), gy(TOTROWS), 1), 256, 0, stream>>>(
      xb, DMODEL, 0, wkb, DMODEL, 0, Kf, nullptr, DMODEL, 1, 0, 0,
      TOTROWS, DMODEL, DMODEL, nullptr, nullptr);
  gemm_bf16_kernel<1><<<dim3(gx(DMODEL), gy(TOTROWS), 1), 256, 0, stream>>>(
      xb, DMODEL, 0, wvb, DMODEL, 0, nullptr, vb, DMODEL, 1, 0, 0,
      TOTROWS, DMODEL, DMODEL, nullptr, nullptr);

  // V^T per (b,h) for attn@V B-operand
  vtrans_kernel<<<dim3((unsigned)((long)TOTROWS * DMODEL / 256)), 256, 0, stream>>>(vb, vtb);

  // fused rel-pos MLP + score  (raw scores -> attn region of d_out)
  score_kernel<<<dim3(SEQ, BB), 256, 0, stream>>>(
      pos, rp_w1, rp_b1, rpw2b, rp_b2, Qf, Kf, attn);

  // softmax over i (rows of 512): fp32 in place + bf16 copy
  softmax_kernel<<<dim3((unsigned)(ATTN_ELEMS / SEQ)), 256, 0, stream>>>(attn, attnb);

  // out_heads[b,j,h*64+d] = attn[b,h,j,:] @ v[b,h,:,d]   (bf16, merged heads)
  gemm_bf16_kernel<1><<<dim3(gx(DHEAD), gy(SEQ), BB * NHEAD), 256, 0, stream>>>(
      attnb, SEQ, (long)SEQ * SEQ, vtb, SEQ, (long)DHEAD * SEQ,
      nullptr, outh_b, DMODEL, NHEAD, (long)SEQ * DMODEL, DHEAD,
      SEQ, DHEAD, SEQ, nullptr, nullptr);

  // fc + residual(enc) -> tmp1 (fp32)
  gemm_bf16_kernel<3><<<dim3(gx(DMODEL), gy(TOTROWS), 1), 256, 0, stream>>>(
      outh_b, DMODEL, 0, wfcb, DMODEL, 0, tmp1, nullptr, DMODEL, 1, 0, 0,
      TOTROWS, DMODEL, DMODEL, nullptr, enc);

  // LN1 -> out1f (fp32) + out1b (bf16)
  layernorm_kernel<<<dim3(TOTROWS), 256, 0, stream>>>(tmp1, ln1_g, ln1_b, out1f, out1b);

  // FFN1: H = relu(out1 @ W1^T + b1) -> bf16
  gemm_bf16_kernel<2><<<dim3(gx(DINNER), gy(TOTROWS), 1), 256, 0, stream>>>(
      out1b, DMODEL, 0, w1b, DMODEL, 0, nullptr, Hb, DINNER, 1, 0, 0,
      TOTROWS, DINNER, DMODEL, ffn_b1, nullptr);

  // FFN2: tmp2 = H @ W2^T + b2 + out1
  gemm_bf16_kernel<4><<<dim3(gx(DMODEL), gy(TOTROWS), 1), 256, 0, stream>>>(
      Hb, DINNER, 0, w2b, DINNER, 0, tmp2, nullptr, DMODEL, 1, 0, 0,
      TOTROWS, DMODEL, DINNER, ffn_b2, out1f);

  // LN2 -> out2 region of d_out
  layernorm_kernel<<<dim3(TOTROWS), 256, 0, stream>>>(tmp2, ln2_g, ln2_b, out2, nullptr);
}
